// ONNXInpaintGeneratorWithMemory_52621939311206
// MI455X (gfx1250) — matrix operand order repaired and checked by reading, not compiled, not run on hardware
//
#include <hip/hip_runtime.h>
#include <hip/hip_bf16.h>

// ---------------------------------------------------------------------------
// Problem constants (after dead-code elimination of the reference: only t=0
// of the spatial block and xc[-1]==x of the temporal block affect outputs;
// the `memory` input is mathematically unused).
// ---------------------------------------------------------------------------
#define HIDN  512
#define NTOK  1620          // H*W tokens
#define MP    1664          // NTOK padded to multiple of 64
#define FFD   2048
#define DH    128
#define NHEAD 4
#define LQ    405           // temporal group length
#define LP    448           // LQ padded to multiple of 64
#define GHN   16            // NG*NH temporal (group,head) pairs

typedef unsigned int u32;
typedef __attribute__((ext_vector_type(16))) __bf16 v16bf;
typedef __attribute__((ext_vector_type(8)))  float  v8f;
typedef __attribute__((ext_vector_type(4)))  u32    u32x4;

union FragU { u32x4 q[2]; v16bf v; };
static_assert(sizeof(FragU) == 32, "frag size");

// A-fragment, 16x32 bf16 (MxK), per CDNA5 ISA 7.12.2:
// lanes 0-15: row m=lane, K pairs {0..7,16..23}; lanes 16-31: row m=lane-16,
// K pairs {8..15,24..31}. Both halves are contiguous 16B in row-major memory.
__device__ __forceinline__ v16bf load_afrag(const __bf16* base, int lda, int lane) {
  int m    = lane & 15;
  int kofs = (lane >> 4) * 8;
  const __bf16* p = base + (long)m * lda + kofs;
  FragU f;
  f.q[0] = *reinterpret_cast<const u32x4*>(p);
  f.q[1] = *reinterpret_cast<const u32x4*>(p + 16);
  return f.v;
}
// B-fragment, 32x16 bf16 (KxN), loaded from BT (N-major, K contiguous):
// lane = column n (mod 16); lanes 0-15 K {0..7,16..23}, lanes 16-31 K {8..15,24..31}.
__device__ __forceinline__ v16bf load_bfrag(const __bf16* baseT, int ldb, int lane) {
  int n    = lane & 15;
  int kofs = (lane >> 4) * 8;
  const __bf16* p = baseT + (long)n * ldb + kofs;
  FragU f;
  f.q[0] = *reinterpret_cast<const u32x4*>(p);
  f.q[1] = *reinterpret_cast<const u32x4*>(p + 16);
  return f.v;
}

// C/D 16x16 f32 layout: VGPR v, lane l -> m = (l>=16)*8 + v, n = l&15.
__device__ __forceinline__ void store_tile(
    v8f c, int m0, int n0, int lane,
    const float* bias, const float* resid, int ldr,
    float* outF, int ldof, __bf16* outB, int ldob, __bf16* outBT, int ldobt,
    int mValid, float scale, int relu)
{
  int n  = n0 + (lane & 15);
  int mo = m0 + (lane >> 4) * 8;
  float bv = bias ? bias[n] : 0.f;
#pragma unroll
  for (int v = 0; v < 8; ++v) {
    int m = mo + v;
    if (m >= mValid) continue;
    float val = c[v] * scale + bv;
    if (relu) val = fmaxf(val, 0.f);
    if (resid) val += resid[(long)m * ldr + n];
    if (outF)  outF [(long)m * ldof  + n] = val;
    if (outB)  outB [(long)m * ldob  + n] = (__bf16)val;
    if (outBT) outBT[(long)n * ldobt + m] = (__bf16)val;
  }
}

// Generic batched bf16 GEMM: C[z] = A[z] (MxK, row-major) * BT[z]^T (BT is NxK)
// Epilogue: *scale, +bias, relu, +residual; store f32 / bf16 / bf16-transposed.
// 128 threads = 4 waves; 64x64 tile per block; 32x32 (2x2 WMMA) per wave.
__global__ __launch_bounds__(128)
void gemm_bf16_kernel(const __bf16* __restrict__ A, int lda, long sA,
                      const __bf16* __restrict__ BT, int ldb, long sB,
                      const float* __restrict__ bias,
                      const float* __restrict__ resid, int ldr,
                      float* __restrict__ outF, int ldof, long sOF,
                      __bf16* __restrict__ outB, int ldob, long sOB,
                      __bf16* __restrict__ outBT, int ldobt,
                      int K, int mValid, float scale, int relu)
{
  const int z = blockIdx.z;
  A  += (long)z * sA;
  BT += (long)z * sB;
  if (outF) outF += (long)z * sOF;
  if (outB) outB += (long)z * sOB;

  const int lane  = threadIdx.x & 31;
  const int wave  = threadIdx.x >> 5;
  const int mBase = blockIdx.x * 64 + (wave >> 1) * 32;
  const int nBase = blockIdx.y * 64 + (wave & 1) * 32;

  v8f c00 = {}, c01 = {}, c10 = {}, c11 = {};
  for (int k0 = 0; k0 < K; k0 += 32) {
    v16bf a0 = load_afrag(A + (long)(mBase +  0) * lda + k0, lda, lane);
    v16bf a1 = load_afrag(A + (long)(mBase + 16) * lda + k0, lda, lane);
    v16bf b0 = load_bfrag(BT + (long)(nBase +  0) * ldb + k0, ldb, lane);
    v16bf b1 = load_bfrag(BT + (long)(nBase + 16) * ldb + k0, ldb, lane);
    c00 = __builtin_amdgcn_wmma_f32_16x16x32_bf16(false, a0, false, b0, (short)0, c00, false, false);
    c01 = __builtin_amdgcn_wmma_f32_16x16x32_bf16(false, a0, false, b1, (short)0, c01, false, false);
    c10 = __builtin_amdgcn_wmma_f32_16x16x32_bf16(false, a1, false, b0, (short)0, c10, false, false);
    c11 = __builtin_amdgcn_wmma_f32_16x16x32_bf16(false, a1, false, b1, (short)0, c11, false, false);
  }
  store_tile(c00, mBase,      nBase,      lane, bias, resid, ldr, outF, ldof, outB, ldob, outBT, ldobt, mValid, scale, relu);
  store_tile(c01, mBase,      nBase + 16, lane, bias, resid, ldr, outF, ldof, outB, ldob, outBT, ldobt, mValid, scale, relu);
  store_tile(c10, mBase + 16, nBase,      lane, bias, resid, ldr, outF, ldof, outB, ldob, outBT, ldobt, mValid, scale, relu);
  store_tile(c11, mBase + 16, nBase + 16, lane, bias, resid, ldr, outF, ldof, outB, ldob, outBT, ldobt, mValid, scale, relu);
}

// Row LayerNorm (512 cols) -> bf16.  Rows >= mValid write zeros (padding).
__global__ __launch_bounds__(128)
void layernorm_kernel(const float* __restrict__ in, const float* __restrict__ g,
                      const float* __restrict__ b, __bf16* __restrict__ out, int mValid)
{
  const int row = blockIdx.x;
  const int tid = threadIdx.x;
  __bf16* orow = out + (long)row * HIDN;
  if (row >= mValid) {
    for (int i = tid; i < HIDN; i += 128) orow[i] = (__bf16)0.f;
    return;
  }
  const float* r = in + (long)row * HIDN;
  float4 xv = reinterpret_cast<const float4*>(r)[tid];
  float s  = xv.x + xv.y + xv.z + xv.w;
  float ss = xv.x * xv.x + xv.y * xv.y + xv.z * xv.z + xv.w * xv.w;
#pragma unroll
  for (int o = 16; o > 0; o >>= 1) { s += __shfl_down(s, o); ss += __shfl_down(ss, o); }
  __shared__ float red[8];
  if ((tid & 31) == 0) { red[(tid >> 5) * 2] = s; red[(tid >> 5) * 2 + 1] = ss; }
  __syncthreads();
  float S  = red[0] + red[2] + red[4] + red[6];
  float SS = red[1] + red[3] + red[5] + red[7];
  float mean = S * (1.f / HIDN);
  float var  = SS * (1.f / HIDN) - mean * mean;
  float inv  = rsqrtf(var + 1e-5f);
  int c = tid * 4;
  orow[c + 0] = (__bf16)((xv.x - mean) * inv * g[c + 0] + b[c + 0]);
  orow[c + 1] = (__bf16)((xv.y - mean) * inv * g[c + 1] + b[c + 1]);
  orow[c + 2] = (__bf16)((xv.z - mean) * inv * g[c + 2] + b[c + 2]);
  orow[c + 3] = (__bf16)((xv.w - mean) * inv * g[c + 3] + b[c + 3]);
}

// In-place masked softmax over bf16 rows; cols >= nvalid become exactly 0.
__global__ __launch_bounds__(256)
void softmax_kernel(__bf16* __restrict__ S, int ld, int nvalid)
{
  __bf16* r = S + (long)blockIdx.x * ld;
  const int tid = threadIdx.x;
  __shared__ float redA[8], redB[8];
  float mx = -3.0e38f;
  for (int i = tid; i < nvalid; i += 256) mx = fmaxf(mx, (float)r[i]);
#pragma unroll
  for (int o = 16; o > 0; o >>= 1) mx = fmaxf(mx, __shfl_down(mx, o));
  if ((tid & 31) == 0) redA[tid >> 5] = mx;
  __syncthreads();
  float m = redA[0];
#pragma unroll
  for (int w = 1; w < 8; ++w) m = fmaxf(m, redA[w]);
  float sum = 0.f;
  for (int i = tid; i < nvalid; i += 256) sum += __expf((float)r[i] - m);
#pragma unroll
  for (int o = 16; o > 0; o >>= 1) sum += __shfl_down(sum, o);
  if ((tid & 31) == 0) redB[tid >> 5] = sum;
  __syncthreads();
  float tot = redB[0] + redB[1] + redB[2] + redB[3] + redB[4] + redB[5] + redB[6] + redB[7];
  float inv = 1.f / tot;
  for (int i = tid; i < ld; i += 256) {
    float v = (i < nvalid) ? __expf((float)r[i] - m) * inv : 0.f;
    r[i] = (__bf16)v;
  }
}

// x -> padded f32 XCf (pad rows zero), and x -> mem_fin[0] copy.
__global__ void copy_pad_kernel(const float* __restrict__ x, float* __restrict__ XCf,
                                float* __restrict__ mem0)
{
  int idx = blockIdx.x * 256 + threadIdx.x;      // exact grid MP*HIDN/256
  int n = idx >> 9;
  float v = 0.f;
  if (n < NTOK) { v = x[idx]; mem0[idx] = v; }
  XCf[idx] = v;
}

__global__ void copy_kernel(const float* __restrict__ src, float* __restrict__ dst, int count)
{
  int i = blockIdx.x * 256 + threadIdx.x;
  if (i < count) dst[i] = src[i];
}

// Transpose + convert weight (Kin x Nout, row-major) -> WT (Nout x Kin) bf16.
__global__ void pack_wt_kernel(const float* __restrict__ W, __bf16* __restrict__ WT,
                               int Kin, int Nout)
{
  long idx = (long)blockIdx.x * 256 + threadIdx.x;
  if (idx >= (long)Kin * Nout) return;
  int i = (int)(idx % Kin);
  int o = (int)(idx / Kin);
  WT[idx] = (__bf16)W[(long)i * Nout + o];
}

// Temporal quadrant gather: XT[gh][l][d] and XTT[gh][d][l] from original x.
// gh = g*NHEAD + h, g = a*2 + b (quadrant), l = i*27 + j, token = (a*15+i)*54 + b*27 + j.
__global__ void gather_xt_kernel(const float* __restrict__ x, __bf16* __restrict__ XT,
                                 __bf16* __restrict__ XTT)
{
  int idx = blockIdx.x * 256 + threadIdx.x;      // exact grid GHN*LP*DH/256
  int d  = idx & 127;
  int l  = (idx >> 7) % LP;
  int gh = idx / (LP * DH);
  int g = gh >> 2, h = gh & 3;
  float v = 0.f;
  if (l < LQ) {
    int a = g >> 1, bq = g & 1;
    int i = l / 27, j = l % 27;
    int n = (a * 15 + i) * 54 + bq * 27 + j;
    v = x[(long)n * HIDN + h * DH + d];
  }
  XT[idx] = (__bf16)v;
  XTT[((long)gh * DH + d) * LP + l] = (__bf16)v;
}

// o_mean[l][h*128+d] = mean over g of OT[(g*NHEAD+h)][l][d]
__global__ void reduce_mean_kernel(const float* __restrict__ OT, float* __restrict__ OMEAN)
{
  int idx = blockIdx.x * 256 + threadIdx.x;
  if (idx >= LQ * HIDN) return;
  int c = idx & 511, l = idx >> 9;
  int h = c >> 7, d = c & 127;
  float s = 0.f;
#pragma unroll
  for (int g = 0; g < 4; ++g) s += OT[((long)(g * NHEAD + h) * LP + l) * DH + d];
  OMEAN[idx] = 0.25f * s;
}

// xu2[n] = x_sp[n] + o_mean[n % 405]   (reference's tile() semantics)
__global__ void add_tile_kernel(const float* __restrict__ XA, const float* __restrict__ OMEAN,
                                float* __restrict__ XU2)
{
  int idx = blockIdx.x * 256 + threadIdx.x;      // exact grid MP*HIDN/256
  int n = idx >> 9, c = idx & 511;
  XU2[idx] = XA[idx] + OMEAN[(n % LQ) * HIDN + c];
}

extern "C" void kernel_launch(void* const* d_in, const int* in_sizes, int n_in,
                              void* d_out, int out_size, void* d_ws, size_t ws_size,
                              hipStream_t stream)
{
  (void)in_sizes; (void)n_in; (void)out_size; (void)ws_size;
  const float* x     = (const float*)d_in[0];
  // d_in[1] (memory) provably does not affect any output.
  const float* s_g1  = (const float*)d_in[2];
  const float* s_b1  = (const float*)d_in[3];
  const float* s_wq  = (const float*)d_in[4];
  const float* s_bq  = (const float*)d_in[5];
  const float* s_wk  = (const float*)d_in[6];
  const float* s_bk  = (const float*)d_in[7];
  const float* s_wv  = (const float*)d_in[8];
  const float* s_bv  = (const float*)d_in[9];
  const float* s_wo  = (const float*)d_in[10];
  const float* s_bo  = (const float*)d_in[11];
  const float* s_g2  = (const float*)d_in[12];
  const float* s_b2  = (const float*)d_in[13];
  const float* s_w1  = (const float*)d_in[14];
  const float* s_fb1 = (const float*)d_in[15];
  const float* s_w2  = (const float*)d_in[16];
  const float* s_fb2 = (const float*)d_in[17];
  const float* t_g2  = (const float*)d_in[18];
  const float* t_b2  = (const float*)d_in[19];
  const float* t_w1  = (const float*)d_in[20];
  const float* t_fb1 = (const float*)d_in[21];
  const float* t_w2  = (const float*)d_in[22];
  const float* t_fb2 = (const float*)d_in[23];

  float* outp     = (float*)d_out;
  float* out_xfin = outp;                               // x_fin
  float* out_mem0 = outp + (long)NTOK * HIDN;           // mem_fin[0] = x
  float* out_mem1 = outp + 2L * NTOK * HIDN;            // mem_fin[1] = x_sp

  // Bump allocator over workspace (256B aligned).
  char* wp = (char*)d_ws;
  auto alloc = [&](size_t bytes) -> void* {
    void* r = (void*)wp;
    wp += (bytes + 255) & ~(size_t)255;
    return r;
  };
  float*  XCf   = (float*)alloc((size_t)MP * HIDN * 4);
  float*  XU    = (float*)alloc((size_t)MP * HIDN * 4);
  float*  XA    = (float*)alloc((size_t)MP * HIDN * 4);
  float*  XU2   = (float*)alloc((size_t)MP * HIDN * 4);
  float*  OMEAN = (float*)alloc((size_t)LQ * HIDN * 4);
  float*  OT    = (float*)alloc((size_t)GHN * LP * DH * 4);
  __bf16* XN    = (__bf16*)alloc((size_t)MP * HIDN * 2);
  __bf16* Qb    = (__bf16*)alloc((size_t)MP * HIDN * 2);
  __bf16* Kb    = (__bf16*)alloc((size_t)MP * HIDN * 2);
  __bf16* VT    = (__bf16*)alloc((size_t)HIDN * MP * 2);  // [h*128+dh][token]
  __bf16* Ob    = (__bf16*)alloc((size_t)MP * HIDN * 2);
  __bf16* YN    = (__bf16*)alloc((size_t)MP * HIDN * 2);
  __bf16* YN2   = (__bf16*)alloc((size_t)MP * HIDN * 2);
  __bf16* H1    = (__bf16*)alloc((size_t)MP * FFD * 2);
  __bf16* H2    = (__bf16*)alloc((size_t)MP * FFD * 2);
  __bf16* Smat  = (__bf16*)alloc((size_t)NHEAD * MP * MP * 2);
  __bf16* Stmp  = (__bf16*)alloc((size_t)GHN * LP * LP * 2);
  __bf16* XT    = (__bf16*)alloc((size_t)GHN * LP * DH * 2);
  __bf16* XTT   = (__bf16*)alloc((size_t)GHN * DH * LP * 2);
  __bf16* wqT   = (__bf16*)alloc((size_t)HIDN * HIDN * 2);
  __bf16* wkT   = (__bf16*)alloc((size_t)HIDN * HIDN * 2);
  __bf16* wvT   = (__bf16*)alloc((size_t)HIDN * HIDN * 2);
  __bf16* woT   = (__bf16*)alloc((size_t)HIDN * HIDN * 2);
  __bf16* w1T   = (__bf16*)alloc((size_t)FFD * HIDN * 2);
  __bf16* w2T   = (__bf16*)alloc((size_t)HIDN * FFD * 2);
  __bf16* tw1T  = (__bf16*)alloc((size_t)FFD * HIDN * 2);
  __bf16* tw2T  = (__bf16*)alloc((size_t)HIDN * FFD * 2);

  const float iscale = 0.08838834764831845f;   // 1/sqrt(DH)
  dim3 b128(128), b256(256);
  dim3 gMN (MP / 64, HIDN / 64, 1);            // (26, 8)
  dim3 gSc (MP / 64, MP / 64, NHEAD);          // (26, 26, 4)
  dim3 gPV (MP / 64, DH / 64, NHEAD);          // (26, 2, 4)
  dim3 gF1 (MP / 64, FFD / 64, 1);             // (26, 32)
  dim3 gSt (LP / 64, LP / 64, GHN);            // (7, 7, 16)
  dim3 gPVt(LP / 64, DH / 64, GHN);            // (7, 2, 16)

  // --- inputs / weight packing ---------------------------------------------
  copy_pad_kernel<<<MP * HIDN / 256, b256, 0, stream>>>(x, XCf, out_mem0);
  pack_wt_kernel<<<HIDN * HIDN / 256, b256, 0, stream>>>(s_wq, wqT, HIDN, HIDN);
  pack_wt_kernel<<<HIDN * HIDN / 256, b256, 0, stream>>>(s_wk, wkT, HIDN, HIDN);
  pack_wt_kernel<<<HIDN * HIDN / 256, b256, 0, stream>>>(s_wv, wvT, HIDN, HIDN);
  pack_wt_kernel<<<HIDN * HIDN / 256, b256, 0, stream>>>(s_wo, woT, HIDN, HIDN);
  pack_wt_kernel<<<HIDN * FFD / 256, b256, 0, stream>>>(s_w1, w1T, HIDN, FFD);
  pack_wt_kernel<<<HIDN * FFD / 256, b256, 0, stream>>>(s_w2, w2T, FFD, HIDN);
  pack_wt_kernel<<<HIDN * FFD / 256, b256, 0, stream>>>(t_w1, tw1T, HIDN, FFD);
  pack_wt_kernel<<<HIDN * FFD / 256, b256, 0, stream>>>(t_w2, tw2T, FFD, HIDN);

  // --- spatial block (t=0 only) --------------------------------------------
  layernorm_kernel<<<MP, b128, 0, stream>>>(XCf, s_g1, s_b1, XN, NTOK);
  gemm_bf16_kernel<<<gMN, b128, 0, stream>>>(XN, HIDN, 0L, wqT, HIDN, 0L, s_bq, nullptr, 0,
      nullptr, 0, 0L, Qb, HIDN, 0L, nullptr, 0, HIDN, MP, 1.f, 0);
  gemm_bf16_kernel<<<gMN, b128, 0, stream>>>(XN, HIDN, 0L, wkT, HIDN, 0L, s_bk, nullptr, 0,
      nullptr, 0, 0L, Kb, HIDN, 0L, nullptr, 0, HIDN, MP, 1.f, 0);
  gemm_bf16_kernel<<<gMN, b128, 0, stream>>>(XN, HIDN, 0L, wvT, HIDN, 0L, s_bv, nullptr, 0,
      nullptr, 0, 0L, nullptr, 0, 0L, VT, MP, HIDN, MP, 1.f, 0);       // V stored transposed
  // scores[h] = Q_h K_h^T / sqrt(DH)
  gemm_bf16_kernel<<<gSc, b128, 0, stream>>>(Qb, HIDN, (long)DH, Kb, HIDN, (long)DH,
      nullptr, nullptr, 0, nullptr, 0, 0L, Smat, MP, (long)MP * MP, nullptr, 0,
      DH, MP, iscale, 0);
  softmax_kernel<<<NHEAD * MP, b256, 0, stream>>>(Smat, MP, NTOK);
  // O_h = P_h V_h
  gemm_bf16_kernel<<<gPV, b128, 0, stream>>>(Smat, MP, (long)MP * MP, VT, MP, (long)DH * MP,
      nullptr, nullptr, 0, nullptr, 0, 0L, Ob, HIDN, (long)DH, nullptr, 0,
      MP, MP, 1.f, 0);
  // xu = xc + O wo + bo
  gemm_bf16_kernel<<<gMN, b128, 0, stream>>>(Ob, HIDN, 0L, woT, HIDN, 0L, s_bo, XCf, HIDN,
      XU, HIDN, 0L, nullptr, 0, 0L, nullptr, 0, HIDN, MP, 1.f, 0);
  layernorm_kernel<<<MP, b128, 0, stream>>>(XU, s_g2, s_b2, YN, NTOK);
  gemm_bf16_kernel<<<gF1, b128, 0, stream>>>(YN, HIDN, 0L, w1T, HIDN, 0L, s_fb1, nullptr, 0,
      nullptr, 0, 0L, H1, FFD, 0L, nullptr, 0, HIDN, MP, 1.f, 1);      // ReLU
  gemm_bf16_kernel<<<gMN, b128, 0, stream>>>(H1, FFD, 0L, w2T, FFD, 0L, s_fb2, XU, HIDN,
      XA, HIDN, 0L, nullptr, 0, 0L, nullptr, 0, FFD, MP, 1.f, 0);      // x_sp
  copy_kernel<<<NTOK * HIDN / 256, b256, 0, stream>>>(XA, out_mem1, NTOK * HIDN);

  // --- temporal block (query = original x, quadrant-grouped) ---------------
  gather_xt_kernel<<<GHN * LP * DH / 256, b256, 0, stream>>>(x, XT, XTT);
  gemm_bf16_kernel<<<gSt, b128, 0, stream>>>(XT, DH, (long)LP * DH, XT, DH, (long)LP * DH,
      nullptr, nullptr, 0, nullptr, 0, 0L, Stmp, LP, (long)LP * LP, nullptr, 0,
      DH, LP, iscale, 0);
  softmax_kernel<<<GHN * LP, b256, 0, stream>>>(Stmp, LP, LQ);
  gemm_bf16_kernel<<<gPVt, b128, 0, stream>>>(Stmp, LP, (long)LP * LP, XTT, LP, (long)DH * LP,
      nullptr, nullptr, 0, OT, DH, (long)LP * DH, nullptr, 0, 0L, nullptr, 0,
      LP, LP, 1.f, 0);
  reduce_mean_kernel<<<(LQ * HIDN + 255) / 256, b256, 0, stream>>>(OT, OMEAN);
  add_tile_kernel<<<MP * HIDN / 256, b256, 0, stream>>>(XA, OMEAN, XU2);
  layernorm_kernel<<<MP, b128, 0, stream>>>(XU2, t_g2, t_b2, YN2, NTOK);
  gemm_bf16_kernel<<<gF1, b128, 0, stream>>>(YN2, HIDN, 0L, tw1T, HIDN, 0L, t_fb1, nullptr, 0,
      nullptr, 0, 0L, H2, FFD, 0L, nullptr, 0, HIDN, MP, 1.f, 1);      // ReLU
  gemm_bf16_kernel<<<gMN, b128, 0, stream>>>(H2, FFD, 0L, tw2T, FFD, 0L, t_fb2, XU2, HIDN,
      out_xfin, HIDN, 0L, nullptr, 0, 0L, nullptr, 0, FFD, NTOK, 1.f, 0); // x_fin
}
